// TextPolicyNetwork_78331613544505
// MI455X (gfx1250) — matrix operand order repaired and checked by reading, not compiled, not run on hardware
//
#include <hip/hip_runtime.h>
#include <hip/hip_bf16.h>
#include <stdint.h>

// Problem constants (match reference)
#define VOCABN 32000
#define EMBN   256
#define HIDN   512
#define MAXLENN 128
#define BOSN   1
#define EOSN   2
#define BATCHN 64
#define SEQTN  2048

typedef __attribute__((ext_vector_type(16))) __bf16 bf16x16;
typedef __attribute__((ext_vector_type(8)))  __bf16 bf16x8;
typedef __attribute__((ext_vector_type(8)))  float  f32x8;

// ---------------------------------------------------------------------------
// WMMA helpers (gfx1250 CDNA5, wave32).  D = A(16x32 bf16) x B(32x16 bf16) + C
// ---------------------------------------------------------------------------
__device__ __forceinline__ f32x8 wmma_bf16(bf16x16 a, bf16x16 b, f32x8 c) {
  // 8 args: (neg_a, A, neg_b, B, c_mod, C, reuse_a, reuse_b)
  return __builtin_amdgcn_wmma_f32_16x16x32_bf16(
      false, a, false, b, (short)0, c, false, false);
}

// A-matrix 16x32 bf16 fragment: lanes 0-15 hold M=lane with
// K = k0+0..15 contiguous (v[0..15]); lanes 16-31 hold
// K = k0+16..31 of the same rows (matches B).  Source is a bf16 row
// (LDS-staged x tile, or the global bf16 h buffer) -> two 16B vector loads.
__device__ __forceinline__ bf16x16 load_a_frag(const __bf16* row, int k0,
                                               int half) {
  const int b0 = k0 + half * 16;
  bf16x8 lo = *(const bf16x8*)(row + b0);
  bf16x8 hi = *(const bf16x8*)(row + b0 + 8);
  bf16x16 a;
#pragma unroll
  for (int i = 0; i < 8; ++i) { a[i] = lo[i]; a[i + 8] = hi[i]; }
  return a;
}

// B-matrix 32x16 bf16: one column per lane (N = lane&15), lanes 0-15 hold
// K=k0..k0+15 contiguous, lanes 16-31 hold K=k0+16..k0+31 (same k order
// as A, not ISA table order).  Weight rows are row-major [N, K] so this is one aligned 32B load.
__device__ __forceinline__ bf16x16 load_b_bf16(const __bf16* __restrict__ row,
                                               int k0, int half) {
  return *(const bf16x16*)(row + k0 + half * 16);
}

__device__ __forceinline__ float sigmoidf_(float x) {
  return 1.0f / (1.0f + __expf(-x));
}

// ---------------------------------------------------------------------------
// fp32 -> bf16 weight conversion (one-time per call; weights then L2-resident)
// ---------------------------------------------------------------------------
__global__ void f2bf_kernel(const float* __restrict__ s, __bf16* __restrict__ d,
                            int n) {
  for (int i = blockIdx.x * blockDim.x + threadIdx.x; i < n;
       i += gridDim.x * blockDim.x)
    d[i] = (__bf16)s[i];
}

__global__ void init_kernel(float* __restrict__ hA, __bf16* __restrict__ hbfA,
                            int* __restrict__ prev) {
  int i = blockIdx.x * blockDim.x + threadIdx.x;
  if (i < BATCHN * HIDN) { hA[i] = 0.0f; hbfA[i] = (__bf16)0.0f; }
  if (i < BATCHN) prev[i] = BOSN;
}

// ---------------------------------------------------------------------------
// One GRU step: h_out = GRUCell(emb[tok], h_in).  Each wave owns a 16x16
// h-tile (72 v_wmma_f32_16x16x32_bf16).  The 16-row x-tile is shared by all
// 8 waves of the block, so it is staged ONCE into LDS as bf16 and re-read as
// ds_load fragments.  The recurrent h A-fragments come from the bf16 h buffer
// written by the previous step (no per-step converts).  Epilogue writes both
// fp32 h (exact recurrence/epilogue math) and bf16 h (next step's A matrix).
// grid = (HID/16/8, B/16), block = 256 (8 waves)
// ---------------------------------------------------------------------------
__global__ __launch_bounds__(256) void gru_step_kernel(
    const int* __restrict__ tok, int tok_stride, int tok_off,
    const float* __restrict__ emb,
    const __bf16* __restrict__ Wih,   // [3H, EMB] bf16 row-major
    const __bf16* __restrict__ Whh,   // [3H, HID] bf16 row-major
    const float* __restrict__ bih, const float* __restrict__ bhh,
    const float* __restrict__ h_in, const __bf16* __restrict__ hbf_in,
    float* __restrict__ h_out, __bf16* __restrict__ hbf_out,
    const int* __restrict__ lengths, int t) {
  __shared__ __align__(16) __bf16 sx[16 * EMBN];  // 8 KB x-tile, bf16

  const int tid  = threadIdx.x;
  const int wave = tid >> 5;
  const int lane = tid & 31;
  const int half = lane >> 4;
  const int ln   = lane & 15;
  const int nb   = (blockIdx.x * 8 + wave) * 16;  // output column tile
  const int mb   = blockIdx.y * 16;               // batch row tile

  // ---- cooperative x-tile stage: 256 threads cover 16 rows x 256 elems ----
  {
    const int r  = tid >> 4;          // row 0..15
    const int c0 = (tid & 15) * 16;   // 16 consecutive elements
    const int token = tok[(mb + r) * tok_stride + tok_off];
    const float* xr = emb + (size_t)token * EMBN + c0;
    __bf16* dst = sx + r * EMBN + c0;
#pragma unroll
    for (int i = 0; i < 16; ++i) dst[i] = (__bf16)xr[i];
  }
  __syncthreads();

  f32x8 axr{}, axz{}, axn{}, ahr{}, ahz{}, ahn{};

  // ---- input-to-hidden pass: A from LDS x-tile, K = EMB ----
  const __bf16* xrow = sx + ln * EMBN;
  const __bf16* wr = Wih + (size_t)(0 * HIDN + nb + ln) * EMBN;
  const __bf16* wz = Wih + (size_t)(1 * HIDN + nb + ln) * EMBN;
  const __bf16* wn = Wih + (size_t)(2 * HIDN + nb + ln) * EMBN;
#pragma unroll
  for (int k0 = 0; k0 < EMBN; k0 += 32) {
    bf16x16 a = load_a_frag(xrow, k0, half);  // ds_load, shared by 3 gates
    axr = wmma_bf16(a, load_b_bf16(wr, k0, half), axr);
    axz = wmma_bf16(a, load_b_bf16(wz, k0, half), axz);
    axn = wmma_bf16(a, load_b_bf16(wn, k0, half), axn);
  }

  // ---- hidden-to-hidden pass: A from global bf16 h (L2-hot), K = HID ----
  const __bf16* hrow = hbf_in + (size_t)(mb + ln) * HIDN;
  const __bf16* ur = Whh + (size_t)(0 * HIDN + nb + ln) * HIDN;
  const __bf16* uz = Whh + (size_t)(1 * HIDN + nb + ln) * HIDN;
  const __bf16* un = Whh + (size_t)(2 * HIDN + nb + ln) * HIDN;
#pragma unroll
  for (int k0 = 0; k0 < HIDN; k0 += 32) {
    bf16x16 a = load_a_frag(hrow, k0, half);
    ahr = wmma_bf16(a, load_b_bf16(ur, k0, half), ahr);
    ahz = wmma_bf16(a, load_b_bf16(uz, k0, half), ahz);
    ahn = wmma_bf16(a, load_b_bf16(un, k0, half), ahn);
  }

  // ---- GRU gate combine (C/D layout: row = mb+i+8*half, col = nb+ln) ----
  const int col = nb + ln;
  const float b_r  = bih[col] + bhh[col];
  const float b_z  = bih[HIDN + col] + bhh[HIDN + col];
  const float b_in = bih[2 * HIDN + col];
  const float b_hn = bhh[2 * HIDN + col];
#pragma unroll
  for (int i = 0; i < 8; ++i) {
    const int row = mb + i + half * 8;
    const float hprev = h_in[(size_t)row * HIDN + col];
    const float r  = sigmoidf_(axr[i] + ahr[i] + b_r);
    const float z  = sigmoidf_(axz[i] + ahz[i] + b_z);
    const float nn = tanhf(axn[i] + b_in + r * (ahn[i] + b_hn));
    float hnew = (1.0f - z) * nn + z * hprev;
    if (lengths != nullptr && !(t < lengths[row])) hnew = hprev;  // packed-seq
    h_out[(size_t)row * HIDN + col] = hnew;
    hbf_out[(size_t)row * HIDN + col] = (__bf16)hnew;
  }
}

// ---------------------------------------------------------------------------
// logits = h @ out_W.T + out_b      [64,512] x [512,32000]
// A comes straight from the bf16 h buffer (no converts); out_W bf16 is
// L2-resident (32 MB < 192 MB L2) so each step streams it at L2 bandwidth.
// grid = (VOCAB/16/8, B/16), block = 256 (8 waves)
// ---------------------------------------------------------------------------
__global__ __launch_bounds__(256) void logits_kernel(
    const __bf16* __restrict__ hbf, const __bf16* __restrict__ Wo,
    const float* __restrict__ bo, float* __restrict__ logits) {
  const int wave = threadIdx.x >> 5;
  const int lane = threadIdx.x & 31;
  const int half = lane >> 4;
  const int ln   = lane & 15;
  const int nb   = (blockIdx.x * 8 + wave) * 16;
  const int mb   = blockIdx.y * 16;

  f32x8 acc{};
  const __bf16* hrow = hbf + (size_t)(mb + ln) * HIDN;
  const __bf16* wrow = Wo + (size_t)(nb + ln) * HIDN;
#pragma unroll
  for (int k0 = 0; k0 < HIDN; k0 += 32) {
    if (k0 + 64 < HIDN)
      __builtin_prefetch(wrow + k0 + 64, 0, 1);  // global_prefetch_b8
    acc = wmma_bf16(load_a_frag(hrow, k0, half),
                    load_b_bf16(wrow, k0, half), acc);
  }
  const int col = nb + ln;
  const float bias = bo[col];
#pragma unroll
  for (int i = 0; i < 8; ++i) {
    const int row = mb + i + half * 8;
    logits[(size_t)row * VOCABN + col] = acc[i] + bias;
  }
}

// ---------------------------------------------------------------------------
// Threefry2x32 (JAX PRNG) — 20 ARX rounds
// ---------------------------------------------------------------------------
__device__ __forceinline__ uint32_t rotl32_(uint32_t x, int r) {
  return (x << r) | (x >> (32 - r));
}
__device__ __forceinline__ void threefry2x32_(uint32_t k0, uint32_t k1,
                                              uint32_t c0, uint32_t c1,
                                              uint32_t* o0, uint32_t* o1) {
  uint32_t ks0 = k0, ks1 = k1, ks2 = 0x1BD11BDAu ^ k0 ^ k1;
  uint32_t x0 = c0 + ks0, x1 = c1 + ks1;
  const int R[8] = {13, 15, 26, 6, 17, 29, 16, 24};
#pragma unroll
  for (int i = 0; i < 5; ++i) {
#pragma unroll
    for (int j = 0; j < 4; ++j) {
      int r = R[4 * (i & 1) + j];
      x0 += x1; x1 = rotl32_(x1, r); x1 ^= x0;
    }
    uint32_t a = (i + 1) % 3 == 0 ? ks0 : ((i + 1) % 3 == 1 ? ks1 : ks2);
    uint32_t b = (i + 2) % 3 == 0 ? ks0 : ((i + 2) % 3 == 1 ? ks1 : ks2);
    x0 += a;
    x1 += b + (uint32_t)(i + 1);
  }
  *o0 = x0; *o1 = x1;
}

// ---------------------------------------------------------------------------
// Gumbel-argmax categorical sample + log-softmax logprob, one block per row.
// Online softmax (running max / scaled sum) + argmax, LDS tree reduction.
// ---------------------------------------------------------------------------
__global__ __launch_bounds__(256) void sample_kernel(
    const float* __restrict__ logits, int step, int* __restrict__ prev,
    int* __restrict__ actions, float* __restrict__ lps) {
  __shared__ float sm[256], ss[256], sv[256];
  __shared__ int si[256];
  __shared__ uint32_t skey[2];
  const int b = blockIdx.x, tid = threadIdx.x;

  if (tid == 0) {  // step key: split(PRNGKey(42), MAXLEN)[step]
    uint32_t o0, o1;
    threefry2x32_(0u, 42u, (uint32_t)step, (uint32_t)(MAXLENN + step), &o0, &o1);
    skey[0] = o0; skey[1] = o1;
  }
  __syncthreads();
  const uint32_t k0 = skey[0], k1 = skey[1];

  float m = -__builtin_inff(), s = 0.0f, bv = -__builtin_inff();
  int bi = 0;
  const uint32_t halfN = (uint32_t)(BATCHN * VOCABN / 2);
  for (int v = tid; v < VOCABN; v += 256) {
    const float x = logits[(size_t)b * VOCABN + v];
    const float nm = fmaxf(m, x);
    s = s * __expf(m - nm) + __expf(x - nm);
    m = nm;
    const uint32_t j = (uint32_t)(b * VOCABN + v);
    uint32_t o0, o1, bits;
    if (j < halfN) { threefry2x32_(k0, k1, j, halfN + j, &o0, &o1); bits = o0; }
    else           { threefry2x32_(k0, k1, j - halfN, j, &o0, &o1); bits = o1; }
    float u = __uint_as_float((bits >> 9) | 0x3F800000u) - 1.0f;  // [0,1)
    u = fmaxf(u, 1.1754944e-38f);
    const float score = x - __logf(-__logf(u));  // logits + gumbel
    if (score > bv || (score == bv && v < bi)) { bv = score; bi = v; }
  }
  sm[tid] = m; ss[tid] = s; sv[tid] = bv; si[tid] = bi;
  __syncthreads();
  for (int off = 128; off > 0; off >>= 1) {
    if (tid < off) {
      const float m2 = sm[tid + off], s2 = ss[tid + off];
      const float M = fmaxf(sm[tid], m2);
      ss[tid] = ss[tid] * __expf(sm[tid] - M) + s2 * __expf(m2 - M);
      sm[tid] = M;
      const float v2 = sv[tid + off]; const int i2 = si[tid + off];
      if (v2 > sv[tid] || (v2 == sv[tid] && i2 < si[tid])) {
        sv[tid] = v2; si[tid] = i2;
      }
    }
    __syncthreads();
  }
  if (tid == 0) {
    const int tokn = si[0];
    const float lp = logits[(size_t)b * VOCABN + tokn] - sm[0] - __logf(ss[0]);
    prev[b] = tokn;
    actions[b * MAXLENN + step] = tokn;
    lps[b * MAXLENN + step] = lp;
  }
}

// ---------------------------------------------------------------------------
// EOS masking + logprob sum + lengths; write outputs as float (actions,
// log_prob[B,1], action_lengths[B] concatenated flat).
// ---------------------------------------------------------------------------
__global__ void finalize_kernel(const int* __restrict__ actions,
                                const float* __restrict__ lps,
                                float* __restrict__ out) {
  const int b = blockIdx.x * blockDim.x + threadIdx.x;
  if (b >= BATCHN) return;
  int c = 0, len = 0;
  float sum = 0.0f;
  for (int t = 0; t < MAXLENN; ++t) {
    const int a = actions[b * MAXLENN + t];
    c += (a == EOSN);
    const int mask = (c <= 1);
    sum += mask ? lps[b * MAXLENN + t] : 0.0f;
    len += mask;
    out[b * MAXLENN + t] = (float)a;
  }
  out[BATCHN * MAXLENN + b] = sum;
  out[BATCHN * MAXLENN + BATCHN + b] = (float)len;
}

// ---------------------------------------------------------------------------
// Host driver — all launches on `stream`, graph-capture safe.
// Workspace usage ≈ 46.1 MB (44 MiB).
// ---------------------------------------------------------------------------
extern "C" void kernel_launch(void* const* d_in, const int* in_sizes, int n_in,
                              void* d_out, int out_size, void* d_ws,
                              size_t ws_size, hipStream_t stream) {
  (void)in_sizes; (void)n_in; (void)out_size; (void)ws_size;
  const int*   tokens  = (const int*)d_in[0];
  const int*   lengths = (const int*)d_in[1];
  const float* emb     = (const float*)d_in[2];
  const float* eWih    = (const float*)d_in[3];
  const float* eWhh    = (const float*)d_in[4];
  const float* ebih    = (const float*)d_in[5];
  const float* ebhh    = (const float*)d_in[6];
  const float* dWih    = (const float*)d_in[7];
  const float* dWhh    = (const float*)d_in[8];
  const float* dbih    = (const float*)d_in[9];
  const float* dbhh    = (const float*)d_in[10];
  const float* outW    = (const float*)d_in[11];
  const float* outb    = (const float*)d_in[12];

  // Carve workspace (256B aligned slices)
  char* p = (char*)d_ws;
  auto carve = [&](size_t bytes) -> void* {
    void* r = (void*)p;
    p += (bytes + 255) & ~(size_t)255;
    return r;
  };
  __bf16* wWihE = (__bf16*)carve((size_t)3 * HIDN * EMBN * 2);
  __bf16* wWhhE = (__bf16*)carve((size_t)3 * HIDN * HIDN * 2);
  __bf16* wWihD = (__bf16*)carve((size_t)3 * HIDN * EMBN * 2);
  __bf16* wWhhD = (__bf16*)carve((size_t)3 * HIDN * HIDN * 2);
  __bf16* wOut  = (__bf16*)carve((size_t)VOCABN * HIDN * 2);
  float*  hA    = (float*)carve((size_t)BATCHN * HIDN * 4);
  float*  hB    = (float*)carve((size_t)BATCHN * HIDN * 4);
  __bf16* hbfA  = (__bf16*)carve((size_t)BATCHN * HIDN * 2);
  __bf16* hbfB  = (__bf16*)carve((size_t)BATCHN * HIDN * 2);
  float*  lgts  = (float*)carve((size_t)BATCHN * VOCABN * 4);
  int*    prev  = (int*)carve((size_t)BATCHN * 4);
  int*    acts  = (int*)carve((size_t)BATCHN * MAXLENN * 4);
  float*  lpsb  = (float*)carve((size_t)BATCHN * MAXLENN * 4);

  // 1) convert weights fp32 -> bf16 (B-matrices, L2-resident thereafter)
  auto cvt = [&](const float* s, __bf16* d, int n) {
    int blocks = (n + 256 * 8 - 1) / (256 * 8);
    if (blocks > 2048) blocks = 2048;
    if (blocks < 1) blocks = 1;
    f2bf_kernel<<<blocks, 256, 0, stream>>>(s, d, n);
  };
  cvt(eWih, wWihE, 3 * HIDN * EMBN);
  cvt(eWhh, wWhhE, 3 * HIDN * HIDN);
  cvt(dWih, wWihD, 3 * HIDN * EMBN);
  cvt(dWhh, wWhhD, 3 * HIDN * HIDN);
  cvt(outW, wOut, VOCABN * HIDN);

  // 2) init h0 = 0 (fp32 + bf16), prev = BOS
  init_kernel<<<(BATCHN * HIDN + 255) / 256, 256, 0, stream>>>(hA, hbfA, prev);

  // 3) encoder: 2048 serial GRU steps (ping-pong h)
  dim3 ggrid(HIDN / 16 / 8, BATCHN / 16);  // (4,4)
  for (int t = 0; t < SEQTN; ++t) {
    const float* hin   = (t & 1) ? hB : hA;
    float* hout        = (t & 1) ? hA : hB;
    const __bf16* bin  = (t & 1) ? hbfB : hbfA;
    __bf16* bout       = (t & 1) ? hbfA : hbfB;
    gru_step_kernel<<<ggrid, 256, 0, stream>>>(
        tokens, SEQTN, t, emb, wWihE, wWhhE, ebih, ebhh, hin, bin, hout, bout,
        lengths, t);
  }
  // final encoder h is in hA/hbfA (t = 2047 is odd)

  // 4) decoder: 128 steps of GRU -> logits -> sample
  dim3 lgrid(VOCABN / 16 / 8, BATCHN / 16);  // (250,4)
  for (int s = 0; s < MAXLENN; ++s) {
    const float* hin  = (s & 1) ? hB : hA;
    float* hout       = (s & 1) ? hA : hB;
    const __bf16* bin = (s & 1) ? hbfB : hbfA;
    __bf16* bout      = (s & 1) ? hbfA : hbfB;
    gru_step_kernel<<<ggrid, 256, 0, stream>>>(
        prev, 1, 0, emb, wWihD, wWhhD, dbih, dbhh, hin, bin, hout, bout,
        nullptr, 0);
    logits_kernel<<<lgrid, 256, 0, stream>>>(bout, wOut, outb, lgts);
    sample_kernel<<<dim3(BATCHN), 256, 0, stream>>>(lgts, s, prev, acts, lpsb);
  }

  // 5) EOS mask, logprob sum, lengths -> d_out
  finalize_kernel<<<1, 64, 0, stream>>>(acts, lpsb, (float*)d_out);
}
